// GraphConsolidator_34677565948786
// MI455X (gfx1250) — compile-verified, matrix operand order checked by reading, not run on hardware
//
#include <hip/hip_runtime.h>
#include <hip/hip_bf16.h>
#include <math.h>
#include <stdint.h>

typedef __attribute__((ext_vector_type(16))) _Float16 v16h;
typedef __attribute__((ext_vector_type(8)))  _Float16 v8h;
typedef __attribute__((ext_vector_type(8)))  float    v8f;
typedef __attribute__((ext_vector_type(4)))  unsigned int u32x4;
typedef __attribute__((ext_vector_type(8)))  int      i32x8;
typedef __attribute__((ext_vector_type(4)))  int      i32x4;

#define D_MEM   128
#define N_SLOTS 8192
#define N_BATCH 4
#define THRESH  0.65f
#define NCHUNK  (N_SLOTS / 32)

#if __has_builtin(__builtin_amdgcn_tensor_load_to_lds) && \
    __has_builtin(__builtin_amdgcn_s_wait_tensorcnt)
#define GC_HAS_TDM 1
#else
#define GC_HAS_TDM 0
#endif

// ---------------------------------------------------------------------------
// Kernel 1: proj = x @ W^T + b ; L2-normalize ; emit f16 proj (row-major) and
// f16 x transposed per batch (xT[b][d][n]). One block of 128 threads per row.
// ---------------------------------------------------------------------------
__global__ __launch_bounds__(128) void gc_prep_kernel(
    const float* __restrict__ x,      // [B*N][128]
    const float* __restrict__ W,      // [128][128]  (out, in)
    const float* __restrict__ bias,   // [128]
    _Float16* __restrict__ projh,     // [B*N][128]
    _Float16* __restrict__ xT)        // [B][128][N]
{
    const int row = blockIdx.x;            // 0 .. B*N-1
    const int b   = row >> 13;             // /8192
    const int n   = row & (N_SLOTS - 1);
    const int o   = threadIdx.x;           // output feature

    __shared__ float xs[D_MEM];
    __shared__ float red[D_MEM];

    const float* xr = x + (size_t)row * D_MEM;
    xs[o] = xr[o];
    __syncthreads();

    float acc = bias[o];
    const float* wr = W + o * D_MEM;
#pragma unroll 8
    for (int d = 0; d < D_MEM; ++d) acc = fmaf(xs[d], wr[d], acc);

    red[o] = acc * acc;
    __syncthreads();
    for (int s = 64; s > 0; s >>= 1) {
        if (o < s) red[o] += red[o + s];
        __syncthreads();
    }
    const float norm = sqrtf(red[0]);
    const float rinv = 1.0f / fmaxf(norm, 1e-12f);

    projh[(size_t)row * D_MEM + o] = (_Float16)(acc * rinv);
    xT[((size_t)b * D_MEM + o) * N_SLOTS + n] = (_Float16)xs[o];
}

// ---------------------------------------------------------------------------
// TDM: issue a 2D tile DMA global->LDS via Tensor DMA Descriptor (D#).
// Descriptor layout per CDNA5 ISA (all element
// units are data_size = 2 bytes).  Groups 2/3 (and the trailing extended
// group of the 6-arg clang-23 builtin) are zero: plain 2-D tile, no cluster
// multicast, no iteration, no padding.
// ---------------------------------------------------------------------------
#if GC_HAS_TDM
__device__ __forceinline__ void gc_tdm_load_2d(
    const void* gptr, uint32_t lds_addr,
    uint32_t tensor_d0, uint32_t tensor_d1,
    uint32_t tile_d0,   uint32_t tile_d1,
    uint32_t stride0)
{
    const uint64_t ga = (uint64_t)(uintptr_t)gptr;
    u32x4 g0;
    g0[0] = 1u;                                      // count=1 valid, user mode
    g0[1] = lds_addr;                                // lds_addr  (bits 63:32)
    g0[2] = (uint32_t)(ga & 0xffffffffu);            // global_addr[31:0]
    g0[3] = (uint32_t)((ga >> 32) & 0x01ffffffu)     // global_addr[56:32]
          | (2u << 30);                              // type = 2 ("image")
    i32x8 g1;
    g1[0] = (int)(1u << 16);                         // data_size=1 -> 2 bytes
    g1[1] = (int)((tensor_d0 & 0xffffu) << 16);      // tensor_dim0[15:0]
    g1[2] = (int)(((tensor_d0 >> 16) & 0xffffu) |    // tensor_dim0[31:16]
                  ((tensor_d1 & 0xffffu) << 16));    // tensor_dim1[15:0]
    g1[3] = (int)(((tensor_d1 >> 16) & 0xffffu) |    // tensor_dim1[31:16]
                  ((tile_d0 & 0xffffu) << 16));      // tile_dim0
    g1[4] = (int)(tile_d1 & 0xffffu);                // tile_dim1 ; tile_dim2=0
    g1[5] = (int)stride0;                            // tensor_dim0_stride[31:0]
    g1[6] = 0;                                       // stride0[47:32], dim1_stride lo
    g1[7] = 0;
    const i32x4 gz4 = {0, 0, 0, 0};
    const i32x8 gz8 = {0, 0, 0, 0, 0, 0, 0, 0};
    __builtin_amdgcn_tensor_load_to_lds(g0, g1, gz4, gz4, gz8, 0);
}
#endif

// ---------------------------------------------------------------------------
// Kernel 2: fused  sim -> threshold -> (adj @ x , deg) -> divide.
// Block = 256 threads (8 waves). Each wave owns a 16-row n-tile; 32-row
// m-chunks are staged in double-buffered LDS by the Tensor Data Mover and
// shared by all 8 waves (DMA of chunk mc+1 overlaps WMMA on chunk mc).
// Per m-chunk per wave: 8 WMMA (sim) + 8 WMMA (acc) + 1 WMMA (deg, B=ones).
// ---------------------------------------------------------------------------
__global__ __launch_bounds__(256) void gc_main_kernel(
    const _Float16* __restrict__ projh,  // [B*N][128]
    const _Float16* __restrict__ xT,     // [B][128][N]
    float* __restrict__ out)             // [B*N][128]
{
    const int lane   = threadIdx.x & 31;
    const int wave   = threadIdx.x >> 5;        // 0..7
    const int batch  = blockIdx.x >> 6;         // 64 blocks per batch
    const int nblk   = blockIdx.x & 63;
    const int n_base = nblk * 128 + wave * 16;  // 16-row tile for this wave

    __shared__ _Float16 s_proj[2][32 * D_MEM];  // m-chunk proj rows [32][128]
    __shared__ _Float16 s_xT[2][D_MEM * 32];    // xT stage [128][32]
    __shared__ _Float16 s_adj[8][16 * 32];      // per-wave adjacency [16][32]

    const _Float16* projB = projh + (size_t)batch * N_SLOTS * D_MEM;
    const _Float16* xTB   = xT    + (size_t)batch * D_MEM * N_SLOTS;

    const int colN = lane & 15;
    const int kb   = (lane >> 4) * 8;

    // Hold this wave's proj A-fragments (4 k-chunks of 32 over D=128) in VGPRs.
    v16h afrag[4];
#pragma unroll
    for (int kc = 0; kc < 4; ++kc) {
        const _Float16* p = projB + ((size_t)n_base + colN) * D_MEM + kc * 32 + kb;
        v8h lo = *(const v8h*)(p);
        v8h hi = *(const v8h*)(p + 16);
#pragma unroll
        for (int i = 0; i < 8; ++i) { afrag[kc][i] = lo[i]; afrag[kc][8 + i] = hi[i]; }
    }

    v16h ones;
#pragma unroll
    for (int i = 0; i < 16; ++i) ones[i] = (_Float16)1.0f;

    const v8f zero = {};
    v8f acc[8];
#pragma unroll
    for (int t = 0; t < 8; ++t) acc[t] = zero;
    v8f deg = zero;

#if GC_HAS_TDM
    const uint32_t lds_proj[2] = { (uint32_t)(uintptr_t)&s_proj[0][0],
                                   (uint32_t)(uintptr_t)&s_proj[1][0] };
    const uint32_t lds_xt[2]   = { (uint32_t)(uintptr_t)&s_xT[0][0],
                                   (uint32_t)(uintptr_t)&s_xT[1][0] };
    // Prologue: DMA chunk 0 into buffer 0 (wave 0 issues; EXEC-independent).
    if (threadIdx.x < 32) {
        gc_tdm_load_2d(projB, lds_proj[0], D_MEM, N_SLOTS, D_MEM, 32, D_MEM);
        gc_tdm_load_2d(xTB,   lds_xt[0],   N_SLOTS, D_MEM, 32, D_MEM, N_SLOTS);
    }
#endif

    for (int mc = 0; mc < NCHUNK; ++mc) {
        const int cur = mc & 1;
#if GC_HAS_TDM
        if (threadIdx.x < 32) __builtin_amdgcn_s_wait_tensorcnt(0); // chunk mc landed
        __syncthreads();   // all waves: chunk mc visible; buf[1-cur] free
        if ((threadIdx.x < 32) && (mc + 1 < NCHUNK)) {
            const int m1 = (mc + 1) * 32;
            gc_tdm_load_2d(projB + (size_t)m1 * D_MEM, lds_proj[1 - cur],
                           D_MEM, N_SLOTS, D_MEM, 32, D_MEM);
            gc_tdm_load_2d(xTB + m1, lds_xt[1 - cur],
                           N_SLOTS, D_MEM, 32, D_MEM, N_SLOTS);
        }
#else
        const int m0 = mc * 32;
        __syncthreads();   // previous stage fully consumed
        {
            const int t = threadIdx.x;
            const v8h* src = (const v8h*)(projB + (size_t)m0 * D_MEM);
            v8h* dst = (v8h*)&s_proj[cur][0];
            dst[t]       = src[t];
            dst[t + 256] = src[t + 256];
            const int d = t >> 1, h = t & 1;
            const v8h* s2 = (const v8h*)(xTB + (size_t)d * N_SLOTS + m0 + h * 16);
            v8h* d2 = (v8h*)(&s_xT[cur][0] + d * 32 + h * 16);
            d2[0] = s2[0];
            d2[1] = s2[1];
        }
        __syncthreads();
#endif

        // ---- sim tiles (2 sub-tiles of 16 m-cols), threshold into s_adj ----
#pragma unroll
        for (int sub = 0; sub < 2; ++sub) {
            v8f sim = zero;
#pragma unroll
            for (int kc = 0; kc < 4; ++kc) {
                const _Float16* p = &s_proj[cur][(sub * 16 + colN) * D_MEM + kc * 32 + kb];
                v8h lo = *(const v8h*)(p);
                v8h hi = *(const v8h*)(p + 16);
                v16h bfrag;
#pragma unroll
                for (int i = 0; i < 8; ++i) { bfrag[i] = lo[i]; bfrag[8 + i] = hi[i]; }
                sim = __builtin_amdgcn_wmma_f32_16x16x32_f16(
                          false, afrag[kc], false, bfrag, (short)0, sim, false, false);
            }
#pragma unroll
            for (int r = 0; r < 8; ++r) {
                const int Mrow = r + 8 * (lane >> 4);
                s_adj[wave][Mrow * 32 + sub * 16 + colN] =
                    (sim[r] > THRESH) ? (_Float16)1.0f : (_Float16)0.0f;
            }
        }

        // ---- adj A-fragment (16n x 32m) from per-wave LDS region ----
        v16h adjfrag;
        {
            const _Float16* p = &s_adj[wave][colN * 32 + kb];
            v8h lo = *(const v8h*)(p);
            v8h hi = *(const v8h*)(p + 16);
#pragma unroll
            for (int i = 0; i < 8; ++i) { adjfrag[i] = lo[i]; adjfrag[8 + i] = hi[i]; }
        }

        // ---- acc[t] += adj @ x  (8 d-tiles) ; deg += adj @ ones ----
#pragma unroll
        for (int t = 0; t < 8; ++t) {
            const _Float16* p = &s_xT[cur][(t * 16 + colN) * 32 + kb];
            v8h lo = *(const v8h*)(p);
            v8h hi = *(const v8h*)(p + 16);
            v16h bfrag;
#pragma unroll
            for (int i = 0; i < 8; ++i) { bfrag[i] = lo[i]; bfrag[8 + i] = hi[i]; }
            acc[t] = __builtin_amdgcn_wmma_f32_16x16x32_f16(
                         false, adjfrag, false, bfrag, (short)0, acc[t], false, false);
        }
        deg = __builtin_amdgcn_wmma_f32_16x16x32_f16(
                  false, adjfrag, false, ones, (short)0, deg, false, false);
    }

    // ---- epilogue: out[n][d] = acc / max(deg, 1) ----
    float* ob = out + ((size_t)batch * N_SLOTS + n_base) * D_MEM;
#pragma unroll
    for (int r = 0; r < 8; ++r) {
        const int Mrow = r + 8 * (lane >> 4);
        const float rinv = 1.0f / fmaxf(deg[r], 1.0f);
#pragma unroll
        for (int t = 0; t < 8; ++t)
            ob[(size_t)Mrow * D_MEM + t * 16 + colN] = acc[t][r] * rinv;
    }
}

// ---------------------------------------------------------------------------
extern "C" void kernel_launch(void* const* d_in, const int* in_sizes, int n_in,
                              void* d_out, int out_size, void* d_ws, size_t ws_size,
                              hipStream_t stream)
{
    const float* x    = (const float*)d_in[0];   // episodic_slots [4,8192,128] f32
    const float* W    = (const float*)d_in[1];   // [128,128]
    const float* bias = (const float*)d_in[2];   // [128]
    float* out        = (float*)d_out;           // [4,8192,128] f32

    // workspace: 8 MB f16 proj + 8 MB f16 xT
    _Float16* projh = (_Float16*)d_ws;
    _Float16* xT    = (_Float16*)((char*)d_ws +
                        (size_t)N_BATCH * N_SLOTS * D_MEM * sizeof(_Float16));

    gc_prep_kernel<<<N_BATCH * N_SLOTS, 128, 0, stream>>>(x, W, bias, projh, xT);
    gc_main_kernel<<<N_BATCH * (N_SLOTS / 128), 256, 0, stream>>>(projh, xT, out);
}
